// DualLSTM_77343771066466
// MI455X (gfx1250) — matrix operand order repaired and checked by reading, not compiled, not run on hardware
//
#include <hip/hip_runtime.h>
#include <hip/hip_bf16.h>
#include <cstdint>

// Problem constants (match reference)
#define T_LEN 2048
#define VOCAB 10000
#define EMB   256
#define HID   512

#define NWG      64     // persistent workgroups for the scan (1 per WGP)
#define SCAN_TPB 512

typedef __attribute__((ext_vector_type(16))) __bf16 v16bf;
typedef __attribute__((ext_vector_type(8)))  float  v8f;

union Frag16 { v16bf v; unsigned long long q[4]; };

__device__ __forceinline__ float sigf(float x) { return 1.0f / (1.0f + __expf(-x)); }

// ---------------------------------------------------------------------------
// Init: zero barrier counters and the h_en/h_cn double buffers.
// ---------------------------------------------------------------------------
__global__ void init_kernel(unsigned* ctrl, float* gh_en, float* gh_cn) {
  int i = threadIdx.x + blockIdx.x * blockDim.x;
  if (i < 2) ctrl[i] = 0u;
  if (i < 2 * HID) { gh_en[i] = 0.0f; gh_cn[i] = 0.0f; }
}

// ---------------------------------------------------------------------------
// Persistent dual-LSTM scan. Each WG owns 8 hidden indices; all its Whh/Wih
// rows live in LDS (fp32) for the whole scan. One grid barrier per timestep.
// ---------------------------------------------------------------------------
__global__ __launch_bounds__(SCAN_TPB)
void lstm_scan_kernel(const int* __restrict__ tok, const int* __restrict__ mask,
                      const float* __restrict__ emb,
                      const float* __restrict__ Wih_en, const float* __restrict__ Whh_en,
                      const float* __restrict__ bih_en, const float* __restrict__ bhh_en,
                      const float* __restrict__ Wih_cn, const float* __restrict__ Whh_cn,
                      const float* __restrict__ bih_cn, const float* __restrict__ bhh_cn,
                      unsigned* ctrl, float* gh_en, float* gh_cn,
                      __bf16* __restrict__ outs_bf)
{
  extern __shared__ float sm[];
  float* Wh    = sm;                    // 64 rows x 512 (rows: net*32+gate*8+hh)
  float* Wi    = Wh + 64 * HID;         // 64 rows x 256
  float* hen_s = Wi + 64 * EMB;         // 512
  float* hcn_s = hen_s + HID;           // 512
  float* xb    = hcn_s + HID;           // 256
  float* pu    = xb + EMB;              // 512 partials (h-dot)
  float* pv    = pu + SCAN_TPB;         // 512 partials (x-dot)
  float* gu    = pv + SCAN_TPB;         // 64
  float* gv    = gu + 64;               // 64

  const int tid   = threadIdx.x;
  const int wg    = blockIdx.x;
  const int hbase = wg * 8;

  // Preload this WG's weight rows into LDS (fp32, stays for the whole scan).
  for (int i = tid; i < 64 * HID; i += SCAN_TPB) {
    int r = i >> 9, k = i & (HID - 1);
    int net = r >> 5, gate = (r >> 3) & 3, hh = r & 7;
    const float* W = net ? Whh_cn : Whh_en;
    Wh[i] = W[(size_t)(gate * HID + hbase + hh) * HID + k];
  }
  for (int i = tid; i < 64 * EMB; i += SCAN_TPB) {
    int r = i >> 8, k = i & (EMB - 1);
    int net = r >> 5, gate = (r >> 3) & 3, hh = r & 7;
    const float* W = net ? Wih_cn : Wih_en;
    Wi[i] = W[(size_t)(gate * HID + hbase + hh) * EMB + k];
  }

  // Owner threads (tid<8) keep cell state + fused biases in registers.
  float ben[4] = {0, 0, 0, 0}, bcn[4] = {0, 0, 0, 0}, creg = 0.0f;
  if (tid < 8) {
    int h = hbase + tid;
    #pragma unroll
    for (int g = 0; g < 4; g++) {
      ben[g] = bih_en[g * HID + h] + bhh_en[g * HID + h];
      bcn[g] = bih_cn[g * HID + h] + bhh_cn[g * HID + h];
    }
  }
  __syncthreads();

  const int task = tid >> 3;     // 0..63 : (net, gate, hh)
  const int sub  = tid & 7;
  const int tnet = task >> 5;

  unsigned* cnt = ctrl;
  unsigned* gen = ctrl + 1;

  for (int t = 0; t < T_LEN; t++) {
    const int buf = t & 1;
    // Stage shared state: h vectors (global, double-buffered) and x = emb[tok].
    hen_s[tid] = gh_en[buf * HID + tid];
    hcn_s[tid] = gh_cn[buf * HID + tid];
    if (tid < EMB) xb[tid] = emb[(size_t)tok[t] * EMB + tid];
    __syncthreads();

    // 64 tasks x 8 lanes: u = h . Whh_row (len 512), v = x . Wih_row (len 256)
    const float* hv  = tnet ? hcn_s : hen_s;
    const float* wr  = &Wh[task * HID];
    const float* wir = &Wi[task * EMB];
    float su = 0.0f, sv = 0.0f;
    #pragma unroll 8
    for (int k = sub; k < HID; k += 8) su += wr[k] * hv[k];
    #pragma unroll 8
    for (int k = sub; k < EMB; k += 8) sv += wir[k] * xb[k];
    pu[tid] = su; pv[tid] = sv;
    __syncthreads();

    if (tid < 64) {
      float u = 0.0f, v = 0.0f;
      #pragma unroll
      for (int j = 0; j < 8; j++) { u += pu[tid * 8 + j]; v += pv[tid * 8 + j]; }
      gu[tid] = u; gv[tid] = v;
    }
    __syncthreads();

    if (tid < 8) {
      int h = hbase + tid;
      float uen[4], ucn[4], ven[4], vcn[4];
      #pragma unroll
      for (int g = 0; g < 4; g++) {
        uen[g] = gu[g * 8 + tid];      ucn[g] = gu[32 + g * 8 + tid];
        ven[g] = gv[g * 8 + tid];      vcn[g] = gv[32 + g * 8 + tid];
      }
      // Branch A: en(x) then cn(dummy) on c
      float i1 = sigf(uen[0] + ven[0] + ben[0]);
      float f1 = sigf(uen[1] + ven[1] + ben[1]);
      float g1 = tanhf(uen[2] + ven[2] + ben[2]);
      float o1 = sigf(uen[3] + ven[3] + ben[3]);
      float c1 = f1 * creg + i1 * g1;
      float hA_en = o1 * tanhf(c1);
      float i2 = sigf(ucn[0] + bcn[0]);
      float f2 = sigf(ucn[1] + bcn[1]);
      float g2 = tanhf(ucn[2] + bcn[2]);
      float o2 = sigf(ucn[3] + bcn[3]);
      float c2 = f2 * c1 + i2 * g2;
      float hA_cn = o2 * tanhf(c2);
      // Branch B: cn(x) then en(dummy) on c
      float i3 = sigf(ucn[0] + vcn[0] + bcn[0]);
      float f3 = sigf(ucn[1] + vcn[1] + bcn[1]);
      float g3 = tanhf(ucn[2] + vcn[2] + bcn[2]);
      float o3 = sigf(ucn[3] + vcn[3] + bcn[3]);
      float c3 = f3 * creg + i3 * g3;
      float hB_cn = o3 * tanhf(c3);
      float i4 = sigf(uen[0] + ben[0]);
      float f4 = sigf(uen[1] + ben[1]);
      float g4 = tanhf(uen[2] + ben[2]);
      float o4 = sigf(uen[3] + ben[3]);
      float c4 = f4 * c3 + i4 * g4;
      float hB_en = o4 * tanhf(c4);

      bool en = mask[t] > 0;
      float he2 = en ? hA_en : hB_en;
      float hc2 = en ? hA_cn : hB_cn;
      creg = en ? c2 : c4;
      gh_en[(buf ^ 1) * HID + h] = he2;
      gh_cn[(buf ^ 1) * HID + h] = hc2;
      outs_bf[(size_t)t * HID + h] = (__bf16)(he2 + hc2);
    }

    // Device-wide barrier (release h writes / acquire before next read).
    __syncthreads();
    if (tid == 0) {
      __threadfence();
      unsigned old = atomicAdd(cnt, 1u);
      if (old == NWG - 1) {
        atomicExch(cnt, 0u);
        __threadfence();
        atomicAdd(gen, 1u);
      } else {
        volatile unsigned* vg = gen;
        unsigned target = (unsigned)(t + 1);
        while (*vg < target) __builtin_amdgcn_s_sleep(2);
      }
      __threadfence();
    }
    __syncthreads();
  }
}

// ---------------------------------------------------------------------------
// WMMA GEMM: C[M,N] = A(bf16)[M,K] * B(f32->bf16)[N,K]^T + bias, optional ReLU.
// WG = 256 thr (8 waves), tile 128x256, K-step 32, wave tile 64x64 (4x4 WMMA).
// blockIdx.x = M tile (fast) so consecutive WGs share the same B tile -> the
// 400MB fc2_W streams from HBM ~once and re-reads hit the 192MB L2.
// A tile staged with global_load_async_to_lds_b128 (ASYNCcnt) when the K-tile
// is full; B staged through VGPRs for the f32->bf16 conversion.
// ---------------------------------------------------------------------------
template <bool RELU_BF16_OUT>
__global__ __launch_bounds__(256)
void gemm_wmma_kernel(const __bf16* __restrict__ A, const float* __restrict__ B,
                      const float* __restrict__ bias, void* __restrict__ Cptr,
                      int M, int N, int K)
{
  constexpr int BM = 128, BN = 256, BK = 32, LDT = BK + 8;  // padded LDS stride
  __shared__ __bf16 As[BM * LDT];
  __shared__ __bf16 Bs[BN * LDT];

  const int tid   = threadIdx.x;
  const int mbase = blockIdx.x * BM;   // M fast-varying across WG ids
  const int nbase = blockIdx.y * BN;
  const int wave  = tid >> 5;
  const int lane  = tid & 31;
  const int lm    = lane & 15;
  const int half  = lane >> 4;
  const int wm    = (wave >> 2) * 64;   // wave row offset in tile
  const int wn    = (wave & 3) * 64;    // wave col offset in tile

  v8f acc[4][4];
  const v8f vzero = {0.f, 0.f, 0.f, 0.f, 0.f, 0.f, 0.f, 0.f};
  #pragma unroll
  for (int i = 0; i < 4; i++)
    #pragma unroll
    for (int j = 0; j < 4; j++) acc[i][j] = vzero;

  for (int k0 = 0; k0 < K; k0 += BK) {
    const bool fullK = (k0 + BK <= K);   // M is always a multiple of BM here
    if (fullK) {
      // Async direct-to-LDS copy of the A tile: 128 rows x 64B, 16B per op.
      #pragma unroll
      for (int q = tid; q < (BM * BK * 2) / 16; q += 256) {   // 512 chunks
        int r = q >> 2, c = q & 3;
        unsigned lds = (unsigned)(uintptr_t)(&As[r * LDT]) + (unsigned)(c * 16);
        uint64_t ga  = (uint64_t)(uintptr_t)(A + (size_t)(mbase + r) * K + k0)
                       + (uint64_t)(c * 16);
        asm volatile("global_load_async_to_lds_b128 %0, %1, off"
                     :: "v"(lds), "v"(ga) : "memory");
      }
    } else {
      #pragma unroll
      for (int i = tid; i < BM * BK; i += 256) {
        int r = i >> 5, c = i & 31;
        int gm = mbase + r, gk = k0 + c;
        __bf16 v = (__bf16)0.0f;
        if (gm < M && gk < K) v = A[(size_t)gm * K + gk];
        As[r * LDT + c] = v;
      }
    }
    // Stage B tile with on-the-fly f32 -> bf16 conversion.
    #pragma unroll
    for (int i = tid; i < BN * BK; i += 256) {
      int r = i >> 5, c = i & 31;
      int gn = nbase + r, gk = k0 + c;
      float v = 0.0f;
      if (gn < N && gk < K) v = B[(size_t)gn * K + gk];
      Bs[r * LDT + c] = (__bf16)v;
    }
    // Prefetch next B K-tile (global_prefetch_b8).
    if (k0 + BK < K) {
      int gn = nbase + tid;
      if (gn < N) __builtin_prefetch(&B[(size_t)gn * K + k0 + BK], 0, 1);
    }
    asm volatile("s_wait_asynccnt 0x0" ::: "memory");
    __syncthreads();

    // A fragments: lanes 0-15 hold K[0..7]/K[16..23], lanes 16-31 K[8..15]/K[24..31]
    Frag16 a[4];
    #pragma unroll
    for (int im = 0; im < 4; im++) {
      const __bf16* row = &As[(wm + im * 16 + lm) * LDT];
      const int kb = half * 8;
      a[im].q[0] = *(const unsigned long long*)(row + kb);
      a[im].q[1] = *(const unsigned long long*)(row + kb + 4);
      a[im].q[2] = *(const unsigned long long*)(row + 16 + kb);
      a[im].q[3] = *(const unsigned long long*)(row + 16 + kb + 4);
    }
    #pragma unroll
    for (int jn = 0; jn < 4; jn++) {
      // B fragment: lane n = lm, lanes 0-15 hold K[0..7]/K[16..23], lanes 16-31 K[8..15]/K[24..31]
      Frag16 b;
      const __bf16* row = &Bs[(wn + jn * 16 + lm) * LDT];
      const int kb = half * 8;
      b.q[0] = *(const unsigned long long*)(row + kb);
      b.q[1] = *(const unsigned long long*)(row + kb + 4);
      b.q[2] = *(const unsigned long long*)(row + 16 + kb);
      b.q[3] = *(const unsigned long long*)(row + 16 + kb + 4);
      #pragma unroll
      for (int im = 0; im < 4; im++)
        acc[im][jn] = __builtin_amdgcn_wmma_f32_16x16x32_bf16(
            false, a[im].v, false, b.v, (short)0, acc[im][jn], false, false);
    }
    __syncthreads();
  }

  // Epilogue: C/D layout -> m = r + 8*half, n = lane&15.
  float*  Cf = (float*)Cptr;
  __bf16* Cb = (__bf16*)Cptr;
  #pragma unroll
  for (int jn = 0; jn < 4; jn++) {
    int gn = nbase + wn + jn * 16 + lm;
    if (gn >= N) continue;
    float bv = bias ? bias[gn] : 0.0f;
    #pragma unroll
    for (int im = 0; im < 4; im++) {
      #pragma unroll
      for (int r = 0; r < 8; r++) {
        int gm = mbase + wm + im * 16 + half * 8 + r;
        if (gm < M) {
          float v = acc[im][jn][r] + bv;
          if (RELU_BF16_OUT) Cb[(size_t)gm * N + gn] = (__bf16)fmaxf(v, 0.0f);
          else               Cf[(size_t)gm * N + gn] = v;
        }
      }
    }
  }
}

// ---------------------------------------------------------------------------
extern "C" void kernel_launch(void* const* d_in, const int* in_sizes, int n_in,
                              void* d_out, int out_size, void* d_ws, size_t ws_size,
                              hipStream_t stream) {
  (void)in_sizes; (void)n_in; (void)out_size; (void)ws_size;
  const int*   tok    = (const int*)d_in[0];
  const int*   mask   = (const int*)d_in[1];
  const float* emb    = (const float*)d_in[2];
  const float* Wih_en = (const float*)d_in[3];
  const float* Whh_en = (const float*)d_in[4];
  const float* bih_en = (const float*)d_in[5];
  const float* bhh_en = (const float*)d_in[6];
  const float* Wih_cn = (const float*)d_in[7];
  const float* Whh_cn = (const float*)d_in[8];
  const float* bih_cn = (const float*)d_in[9];
  const float* bhh_cn = (const float*)d_in[10];
  const float* fc1_W  = (const float*)d_in[11];
  const float* fc1_b  = (const float*)d_in[12];
  const float* fc2_W  = (const float*)d_in[13];
  const float* fc2_b  = (const float*)d_in[14];

  char* ws = (char*)d_ws;
  unsigned* ctrl  = (unsigned*)ws;                                  // 256 B
  float*    gh_en = (float*)(ws + 256);                             // 2*512 f32
  float*    gh_cn = gh_en + 2 * HID;                                // 2*512 f32
  __bf16*   outs_bf = (__bf16*)(ws + 256 + (size_t)4 * HID * 4);    // T*H bf16
  __bf16*   hmid_bf = (__bf16*)((char*)outs_bf + (size_t)T_LEN * HID * 2); // T*V bf16

  init_kernel<<<4, 256, 0, stream>>>(ctrl, gh_en, gh_cn);

  const size_t scanLds =
      (size_t)(64 * HID + 64 * EMB + 2 * HID + EMB + 2 * SCAN_TPB + 128) * sizeof(float);
  lstm_scan_kernel<<<NWG, SCAN_TPB, scanLds, stream>>>(
      tok, mask, emb, Wih_en, Whh_en, bih_en, bhh_en,
      Wih_cn, Whh_cn, bih_cn, bhh_cn, ctrl, gh_en, gh_cn, outs_bf);

  dim3 grid((T_LEN + 127) / 128, (VOCAB + 255) / 256);  // x = M tiles (fast)
  // FC1: outs(bf16) @ fc1_W^T + b, ReLU, bf16 out (feeds FC2)
  gemm_wmma_kernel<true><<<grid, 256, 0, stream>>>(
      outs_bf, fc1_W, fc1_b, hmid_bf, T_LEN, VOCAB, HID);
  // FC2: hmid(bf16) @ fc2_W^T + b, f32 out -> d_out   (0.41 TFLOP, WMMA-bound)
  gemm_wmma_kernel<false><<<grid, 256, 0, stream>>>(
      hmid_bf, fc2_W, fc2_b, d_out, T_LEN, VOCAB, VOCAB);
}
